// GroupLinear_20409684590616
// MI455X (gfx1250) — matrix operand order repaired and checked by reading, not compiled, not run on hardware
//
#include <hip/hip_runtime.h>

// GroupLinear for MI455X (gfx1250, wave32, WMMA + async global->LDS).
// y[n,s,:] = x[n,s,:] @ weight[group[n,s]].T
// Pipeline:
//   1) bucket tokens by group (hist/scan/scatter) into ws
//   2) pre-convert x (gathered into sorted order) and w to hi/lo bf16 pairs
//   3) grouped GEMM: 64x128 block tile, wave owns 2x2 16x16 C tiles,
//      double-buffered GLOBAL_LOAD_ASYNC_TO_LDS_B128 staging (ASYNCcnt),
//      3x v_wmma_f32_16x16x32_bf16 per (A,B) pair (hi/lo split ~ fp32 accuracy)

#define IN_F   1024
#define OUT_F  4096
#define NTOK   8192
#define NGRP   8
#define MT     64            // token tile (4 x 16)
#define NTB    128           // output tile (8 x 16)
#define KC     32            // K chunk per WMMA step
#define KCP    40            // padded LDS row stride (80B)
#define NCH    (IN_F / KC)   // 32 chunks
#define MAX_TILES (NTOK / MT + NGRP)   // 136

#define X_ELEMS ((size_t)NTOK * IN_F)            // 8.39M
#define W_ELEMS ((size_t)NGRP * OUT_F * IN_F)    // 33.55M
#define META_BYTES ((size_t)1 << 16)

typedef __bf16 bf16;
typedef __attribute__((ext_vector_type(16))) __bf16 v16bf;
typedef __attribute__((ext_vector_type(8)))  __bf16 v8bf;
typedef __attribute__((ext_vector_type(8)))  float   v8f;

// ---------- bf16 helpers (round-to-nearest-even) ----------
__device__ __forceinline__ bf16 f2bf(float f) {
  unsigned u = __builtin_bit_cast(unsigned, f);
  unsigned r = (u + 0x7FFFu + ((u >> 16) & 1u)) >> 16;
  unsigned short s = (unsigned short)r;
  return __builtin_bit_cast(bf16, s);
}
__device__ __forceinline__ float bf2f(bf16 b) {
  unsigned short s = __builtin_bit_cast(unsigned short, b);
  unsigned u = ((unsigned)s) << 16;
  return __builtin_bit_cast(float, u);
}
__device__ __forceinline__ void split_store4(bf16* hi, bf16* lo, float4 v) {
  float f0 = v.x, f1 = v.y, f2 = v.z, f3 = v.w;
  bf16 h0 = f2bf(f0), h1 = f2bf(f1), h2 = f2bf(f2), h3 = f2bf(f3);
  bf16 l0 = f2bf(f0 - bf2f(h0)), l1 = f2bf(f1 - bf2f(h1));
  bf16 l2 = f2bf(f2 - bf2f(h2)), l3 = f2bf(f3 - bf2f(h3));
  union P { bf16 b[4]; uint2 u; };
  P ph, pl;
  ph.b[0] = h0; ph.b[1] = h1; ph.b[2] = h2; ph.b[3] = h3;
  pl.b[0] = l0; pl.b[1] = l1; pl.b[2] = l2; pl.b[3] = l3;
  *(uint2*)hi = ph.u;
  *(uint2*)lo = pl.u;
}

// A fragment (16x32 bf16, ISA 7.12.2): lanes 0-15 -> K 0-7 | 16-23,
// lanes 16-31 -> K 8-15 | 24-31.
__device__ __forceinline__ v16bf ldA(const bf16* row, int half) {
  union { v16bf v; v8bf h[2]; } u;
  u.h[0] = *(const v8bf*)(row + 8 * half);
  u.h[1] = *(const v8bf*)(row + 16 + 8 * half);
  return u.v;
}
// B fragment (32x16 bf16): lane = column n; lanes 0-15 -> K 0-7 | 16-23,
// lanes 16-31 -> K 8-15 | 24-31 (LDS holds B transposed: W[o][k]).
__device__ __forceinline__ v16bf ldB(const bf16* row, int half) {
  union { v16bf v; v8bf h[2]; } u;
  u.h[0] = *(const v8bf*)(row + 8 * half);
  u.h[1] = *(const v8bf*)(row + 16 + 8 * half);
  return u.v;
}
#define WMMA_BF16(A, B, C) \
  __builtin_amdgcn_wmma_f32_16x16x32_bf16(false, (A), false, (B), (short)0, (C), false, false)

// CDNA5 async direct global->LDS copy, 16B per lane, tracked by ASYNCcnt.
__device__ __forceinline__ void async_ld16(unsigned lds_off, const bf16* g) {
  unsigned long long ga = (unsigned long long)(uintptr_t)g;
  asm volatile("global_load_async_to_lds_b128 %0, %1, off"
               :: "v"(lds_off), "v"(ga) : "memory");
}
__device__ __forceinline__ unsigned lds_off_of(const void* p) {
  return (unsigned)(uintptr_t)p;   // flat LDS addr low 32 bits = LDS offset
}

// ---------- pass 1: token bucketing ----------
// ws ints: [0..7] counts, [8..16] offsets, [17..24] fill, [32..32+8191] sorted ids
__global__ void gl_init(int* meta) {
  if (threadIdx.x < 32) meta[threadIdx.x] = 0;
}
__global__ void gl_hist(const long long* __restrict__ group, int* meta) {
  int i = blockIdx.x * blockDim.x + threadIdx.x;
  if (i < NTOK) atomicAdd(&meta[(int)group[i]], 1);
}
__global__ void gl_scan(int* meta) {
  if (threadIdx.x == 0) {
    int acc = 0;
    meta[8] = 0;
    for (int g = 0; g < NGRP; ++g) { acc += meta[g]; meta[9 + g] = acc; }
    for (int g = 0; g < NGRP; ++g) meta[17 + g] = meta[8 + g];
  }
}
__global__ void gl_scatter(const long long* __restrict__ group, int* meta) {
  int i = blockIdx.x * blockDim.x + threadIdx.x;
  if (i < NTOK) {
    int g = (int)group[i];
    int p = atomicAdd(&meta[17 + g], 1);
    meta[32 + p] = i;
  }
}

// ---------- pass 2: one-time fp32 -> hi/lo bf16 conversion ----------
__global__ void gl_conv_x(const float* __restrict__ x, const int* __restrict__ meta,
                          bf16* __restrict__ xh, bf16* __restrict__ xl) {
  int i = blockIdx.x * blockDim.x + threadIdx.x;   // one float4 each
  int row = i >> 8;                                // IN_F/4 = 256 quads/row
  int kc  = (i & 255) * 4;
  int tok = meta[32 + row];
  float4 v = *(const float4*)(x + (size_t)tok * IN_F + kc);
  size_t o = (size_t)row * IN_F + kc;
  split_store4(xh + o, xl + o, v);
}
__global__ void gl_conv_w(const float* __restrict__ w,
                          bf16* __restrict__ wh, bf16* __restrict__ wl) {
  size_t i = ((size_t)blockIdx.x * blockDim.x + threadIdx.x) * 4;
  float4 v = *(const float4*)(w + i);
  split_store4(wh + i, wl + i, v);
}

// ---------- pass 3: grouped GEMM ----------
__launch_bounds__(256)
__global__ void gl_gemm(const bf16* __restrict__ xh, const bf16* __restrict__ xl,
                        const bf16* __restrict__ wh, const bf16* __restrict__ wl,
                        const int* __restrict__ meta, float* __restrict__ out) {
  __shared__ bf16 Ah[2][MT][KCP],  Al[2][MT][KCP];
  __shared__ bf16 Bh[2][NTB][KCP], Bl[2][NTB][KCP];

  const int* offs   = meta + 8;
  const int* sorted = meta + 32;

  // map blockIdx.x -> (group, local tile)
  int tile = blockIdx.x;
  int g = -1, local = 0, cnt = 0, base = 0, acc = 0;
  for (int gg = 0; gg < NGRP; ++gg) {
    int c = offs[gg + 1] - offs[gg];
    int t = (c + MT - 1) / MT;
    if (tile < acc + t) { g = gg; local = tile - acc; cnt = c; base = offs[gg]; break; }
    acc += t;
  }
  if (g < 0) return;

  const int tid  = threadIdx.x;
  const int lane = tid & 31;
  const int wave = tid >> 5;
  const int wm   = wave & 1;     // wave covers m-subtiles {2wm, 2wm+1}
  const int wn   = wave >> 1;    // and n-subtiles {2wn, 2wn+1}
  const int row0 = local * MT;
  const int half = lane >> 4;
  const int lm   = lane & 15;

  // async staging assignments (16B per instruction per lane)
  const int a_r  = tid >> 2;            // 0..63, 4 threads/row, 8 bf16 each
  const int a_k8 = (tid & 3) * 8;
  const size_t a_off = (size_t)(base + min(row0 + a_r, cnt - 1)) * IN_F + a_k8;
  const bf16* agh = xh + a_off;
  const bf16* agl = xl + a_off;

  const int b_r   = tid >> 1;           // 0..127, 2 threads/row, 16 bf16 each
  const int b_k16 = (tid & 1) * 16;
  const size_t b_off = (size_t)g * OUT_F * IN_F
                     + (size_t)(blockIdx.y * NTB + b_r) * IN_F + b_k16;
  const bf16* bgh = wh + b_off;
  const bf16* bgl = wl + b_off;

  const unsigned ah_d[2] = { lds_off_of(&Ah[0][a_r][a_k8]), lds_off_of(&Ah[1][a_r][a_k8]) };
  const unsigned al_d[2] = { lds_off_of(&Al[0][a_r][a_k8]), lds_off_of(&Al[1][a_r][a_k8]) };
  const unsigned bh_d[2] = { lds_off_of(&Bh[0][b_r][b_k16]), lds_off_of(&Bh[1][b_r][b_k16]) };
  const unsigned bl_d[2] = { lds_off_of(&Bl[0][b_r][b_k16]), lds_off_of(&Bl[1][b_r][b_k16]) };

  // fragment row pointers (buffer 0; buffer 1 adds the array stride)
  const int sub0 = (wm * 2 + 0) * 16, sub1 = (wm * 2 + 1) * 16;
  const int AS = MT * KCP, BS = NTB * KCP;
  const bf16* arh0 = &Ah[0][sub0 + lm][0];
  const bf16* arh1 = &Ah[0][sub1 + lm][0];
  const bf16* arl0 = &Al[0][sub0 + lm][0];
  const bf16* arl1 = &Al[0][sub1 + lm][0];
  const bf16* brh0 = &Bh[0][(wn * 2 + 0) * 16 + lm][0];
  const bf16* brh1 = &Bh[0][(wn * 2 + 1) * 16 + lm][0];
  const bf16* brl0 = &Bl[0][(wn * 2 + 0) * 16 + lm][0];
  const bf16* brl1 = &Bl[0][(wn * 2 + 1) * 16 + lm][0];

  // hoist epilogue token ids (loop-invariant)
  int tk0[8], tk1[8];
  #pragma unroll
  for (int r = 0; r < 8; ++r) {
    tk0[r] = sorted[base + min(row0 + sub0 + 8 * half + r, cnt - 1)];
    tk1[r] = sorted[base + min(row0 + sub1 + 8 * half + r, cnt - 1)];
  }

  v8f c00 = {}, c01 = {}, c10 = {}, c11 = {};

  // prologue: async-stage chunk 0 into buffer 0
  async_ld16(ah_d[0], agh);
  async_ld16(al_d[0], agl);
  async_ld16(bh_d[0], bgh);
  async_ld16(bh_d[0] + 16, bgh + 8);
  async_ld16(bl_d[0], bgl);
  async_ld16(bl_d[0] + 16, bgl + 8);

  for (int i = 0; i < NCH; ++i) {
    const int p = i & 1;
    if (i + 1 < NCH) {
      const int q = p ^ 1;
      const int kn = (i + 1) * KC;
      async_ld16(ah_d[q], agh + kn);
      async_ld16(al_d[q], agl + kn);
      async_ld16(bh_d[q], bgh + kn);
      async_ld16(bh_d[q] + 16, bgh + kn + 8);
      async_ld16(bl_d[q], bgl + kn);
      async_ld16(bl_d[q] + 16, bgl + kn + 8);
      asm volatile("s_wait_asynccnt 0x6" ::: "memory");  // chunk i landed
    } else {
      asm volatile("s_wait_asynccnt 0x0" ::: "memory");
    }
    __syncthreads();                       // all waves' chunk-i copies visible

    const int po = p ? AS : 0;             // A buffer stride (elements)
    const int pb = p ? BS : 0;             // B buffer stride
    v16bf a0h = ldA(arh0 + po, half), a1h = ldA(arh1 + po, half);
    v16bf a0l = ldA(arl0 + po, half), a1l = ldA(arl1 + po, half);
    v16bf b0h = ldB(brh0 + pb, half), b1h = ldB(brh1 + pb, half);
    v16bf b0l = ldB(brl0 + pb, half), b1l = ldB(brl1 + pb, half);

    // (xh+xl)(wh+wl) ~= xh*wh + xh*wl + xl*wh   (~2^-16 rel error)
    c00 = WMMA_BF16(a0h, b0h, c00);
    c01 = WMMA_BF16(a0h, b1h, c01);
    c10 = WMMA_BF16(a1h, b0h, c10);
    c11 = WMMA_BF16(a1h, b1h, c11);
    c00 = WMMA_BF16(a0h, b0l, c00);
    c01 = WMMA_BF16(a0h, b1l, c01);
    c10 = WMMA_BF16(a1h, b0l, c10);
    c11 = WMMA_BF16(a1h, b1l, c11);
    c00 = WMMA_BF16(a0l, b0h, c00);
    c01 = WMMA_BF16(a0l, b1h, c01);
    c10 = WMMA_BF16(a1l, b0h, c10);
    c11 = WMMA_BF16(a1l, b1h, c11);

    __syncthreads();                       // release buffer p for chunk i+2
  }

  // C layout: VGPR r -> M = r + 8*(lane>>4), N = lane&15
  const int nb  = blockIdx.y * NTB;
  const int nc0 = nb + (wn * 2 + 0) * 16 + lm;
  const int nc1 = nb + (wn * 2 + 1) * 16 + lm;
  #pragma unroll
  for (int r = 0; r < 8; ++r) {
    if (row0 + sub0 + 8 * half + r < cnt) {
      float* o = out + (size_t)tk0[r] * OUT_F;
      o[nc0] = c00[r];
      o[nc1] = c01[r];
    }
    if (row0 + sub1 + 8 * half + r < cnt) {
      float* o = out + (size_t)tk1[r] * OUT_F;
      o[nc0] = c10[r];
      o[nc1] = c11[r];
    }
  }
}

extern "C" void kernel_launch(void* const* d_in, const int* in_sizes, int n_in,
                              void* d_out, int out_size, void* d_ws, size_t ws_size,
                              hipStream_t stream) {
  const float*     x     = (const float*)d_in[0];
  const long long* group = (const long long*)d_in[1];
  const float*     wgt   = (const float*)d_in[2];
  float*           out   = (float*)d_out;

  char* wsb  = (char*)d_ws;
  int*  meta = (int*)wsb;
  bf16* xs_h = (bf16*)(wsb + META_BYTES);
  bf16* xs_l = xs_h + X_ELEMS;
  bf16* w_h  = xs_l + X_ELEMS;
  bf16* w_l  = w_h + W_ELEMS;
  // ws requirement: 64KB + 2*(X_ELEMS + W_ELEMS)*2B ~= 168 MB

  gl_init<<<1, 32, 0, stream>>>(meta);
  gl_hist<<<NTOK / 256, 256, 0, stream>>>(group, meta);
  gl_scan<<<1, 32, 0, stream>>>(meta);
  gl_scatter<<<NTOK / 256, 256, 0, stream>>>(group, meta);

  gl_conv_x<<<(int)(X_ELEMS / 4 / 256), 256, 0, stream>>>(x, meta, xs_h, xs_l);
  gl_conv_w<<<(int)(W_ELEMS / 4 / 256), 256, 0, stream>>>(wgt, w_h, w_l);

  dim3 grid(MAX_TILES, OUT_F / NTB);
  gl_gemm<<<grid, 256, 0, stream>>>(xs_h, xs_l, w_h, w_l, meta, out);
}
